// TrajPreAttnAvgLongUser_62405874811210
// MI455X (gfx1250) — matrix operand order repaired and checked by reading, not compiled, not run on hardware
//
#include <hip/hip_runtime.h>

// ---------------------------------------------------------------------------
// TrajPreAttnAvgLongUser on MI455X (gfx1250).
// bf16 WMMA for all GEMMs + flash attention (fp32 accumulate), serial GRU as a
// persistent single-WG scan (32 waves, split-K, bf16-packed weights).
// Flash attention stages history chunks into LDS with the Tensor Data Mover
// (tensor_load_to_lds, TENSORcnt) double-buffered against compute.
// ---------------------------------------------------------------------------

typedef __bf16 bf16_t;
typedef __attribute__((ext_vector_type(16))) __bf16 v16bf;
typedef __attribute__((ext_vector_type(8)))  __bf16 v8bf;
typedef __attribute__((ext_vector_type(4)))  __bf16 v4bf;
typedef __attribute__((ext_vector_type(8)))  float  v8f;
typedef __attribute__((ext_vector_type(4)))  float  v4f;
typedef __attribute__((ext_vector_type(4)))  unsigned int v4u;
typedef __attribute__((ext_vector_type(8)))  int v8i;
typedef __attribute__((ext_vector_type(4)))  int v4i;

static constexpr int SEQ = 4096, HIS = 32768, TGT = 2048;
static constexpr int LOC_E = 512, TIM_E = 64, HID = 512, UID_E = 64;
static constexpr int IN = LOC_E + TIM_E;          // 576
static constexpr int OUTD = 2 * HID + UID_E;      // 1088

#if __has_builtin(__builtin_amdgcn_tensor_load_to_lds) && \
    __has_builtin(__builtin_amdgcn_s_wait_tensorcnt)
#define FLASH_TDM 1
#else
#define FLASH_TDM 0
#endif

__device__ inline bf16_t f2bf(float f) {
  unsigned u = __builtin_bit_cast(unsigned, f);
  unsigned r = (u + 0x7FFFu + ((u >> 16) & 1u)) >> 16;
  return __builtin_bit_cast(bf16_t, (unsigned short)r);
}

__device__ inline v8f zero8() {
  v8f z;
#pragma unroll
  for (int i = 0; i < 8; ++i) z[i] = 0.f;
  return z;
}

// A fragment (16x32 bf16): lane = row (lane&15), K halves split by lane>>4.
__device__ inline v16bf load_a_frag(const bf16_t* base, int ld, int lane) {
  int m = lane & 15, hi = lane >> 4;
  const bf16_t* p = base + (size_t)m * ld + hi * 8;
  union { v16bf v; v8bf h[2]; } u;
  u.h[0] = *(const v8bf*)(p);
  u.h[1] = *(const v8bf*)(p + 16);
  return u.v;
}

// B fragment (32x16 bf16) where B[k][n] comes from a row-major matrix whose
// ROW index is n: lane = n, K halves split by lane>>4 (same order as A).
__device__ inline v16bf load_b_frag_rows(const bf16_t* base, int ld, int lane) {
  int n = lane & 15, hi = lane >> 4;
  const bf16_t* p = base + (size_t)n * ld + hi * 8;
  union { v16bf v; v8bf h[2]; } u;
  u.h[0] = *(const v8bf*)(p);
  u.h[1] = *(const v8bf*)(p + 16);
  return u.v;
}

__device__ inline v8f wmma_bf16(v16bf a, v16bf b, v8f c) {
  return __builtin_amdgcn_wmma_f32_16x16x32_bf16(false, a, false, b, (short)0, c,
                                                 false, false);
}

#if FLASH_TDM
// Issue a 2D TDM load: tile (tile_x elems contiguous) x (tile_y rows) of
// 2-byte elements from global row-major (row stride = stride_x elems) into
// LDS, with optional LDS row padding so the destination keeps its bank-spread
// stride. pad_interval_code per D# encoding, pad_dwords = DWORDs of padding.
__device__ inline void tdm_load_2d(unsigned lds_addr, const void* gaddr,
                                   unsigned tile_x, unsigned tile_y,
                                   unsigned stride_x, unsigned pad_interval_code,
                                   unsigned pad_dwords) {
  unsigned long long ga = (unsigned long long)gaddr;
  v4u g0;
  g0[0] = 1u;  // count=1, user descriptor
  g0[1] = lds_addr;
  g0[2] = (unsigned)(ga & 0xFFFFFFFFu);
  g0[3] = (unsigned)((ga >> 32) & 0x01FFFFFFu) | (2u << 30);  // type=2 (image)
  unsigned w0 = (1u << 16);  // data_size = 2 bytes
  if (pad_dwords) w0 |= (1u << 20) | (pad_interval_code << 22) | ((pad_dwords - 1u) << 25);
  v8i g1;
  g1[0] = (int)w0;
  g1[1] = (int)((tile_x & 0xFFFFu) << 16);  // tensor_dim0[15:0] (== tile_x)
  g1[2] = (int)(((tile_x >> 16) & 0xFFFFu) | ((tile_y & 0xFFFFu) << 16));  // dim0 hi | dim1 lo
  g1[3] = (int)(((tile_y >> 16) & 0xFFFFu) | ((tile_x & 0xFFFFu) << 16));  // dim1 hi | tile_dim0
  g1[4] = (int)(tile_y & 0xFFFFu);  // tile_dim1 (tile_dim2 = 0)
  g1[5] = (int)stride_x;            // tensor_dim0_stride[31:0]
  g1[6] = 0;
  g1[7] = 0;
  v4i z = {0, 0, 0, 0};
#if __clang_major__ >= 23
  v8i z8 = {0, 0, 0, 0, 0, 0, 0, 0};
  __builtin_amdgcn_tensor_load_to_lds(g0, g1, z, z, z8, 0);
#else
  __builtin_amdgcn_tensor_load_to_lds(g0, g1, z, z, 0);
#endif
}
#endif  // FLASH_TDM

// ---------------------------------------------------------------------------
// Elementwise prep kernels
// ---------------------------------------------------------------------------
__global__ void k_concat_cast(const float* __restrict__ a, const float* __restrict__ b,
                              bf16_t* __restrict__ dst, int rows, int d1, int d2) {
  int idx = blockIdx.x * blockDim.x + threadIdx.x;
  int D = d1 + d2;
  if (idx >= rows * D) return;
  int r = idx / D, c = idx - r * D;
  float v = (c < d1) ? a[(size_t)r * d1 + c] : b[(size_t)r * d2 + (c - d1)];
  dst[idx] = f2bf(v);
}

__global__ void k_cast(const float* __restrict__ src, bf16_t* __restrict__ dst, int n) {
  int idx = blockIdx.x * blockDim.x + threadIdx.x;
  if (idx < n) dst[idx] = f2bf(src[idx]);
}

// Pack W_hh (3*HID x HID, row-major, fp32) into [k][unit][4xbf16] (r,z,n,pad)
// so the GRU scan does ONE coalesced b64 load per (k,unit) instead of 3 b32s.
__global__ void k_pack_whh(const float* __restrict__ w, bf16_t* __restrict__ wpk) {
  int idx = blockIdx.x * blockDim.x + threadIdx.x;
  if (idx >= HID * HID) return;
  int k = idx >> 9, u = idx & (HID - 1);
  size_t o = ((size_t)k * HID + u) * 4;
  wpk[o + 0] = f2bf(w[(size_t)(0 * HID + u) * HID + k]);
  wpk[o + 1] = f2bf(w[(size_t)(1 * HID + u) * HID + k]);
  wpk[o + 2] = f2bf(w[(size_t)(2 * HID + u) * HID + k]);
  wpk[o + 3] = f2bf(0.f);
}

__global__ void k_uid_embed(const int* __restrict__ uid, const float* __restrict__ emb,
                            bf16_t* __restrict__ outb) {
  int idx = blockIdx.x * blockDim.x + threadIdx.x;
  if (idx >= TGT * UID_E) return;
  int i = idx >> 6, e = idx & 63;
  outb[(size_t)i * OUTD + 2 * HID + e] = f2bf(emb[(size_t)uid[i] * UID_E + e]);
}

// ---------------------------------------------------------------------------
// Generic WMMA GEMM:  C[M,N] = act(A[M,K] @ W[N,K]^T + bias[N])
// Block = 4 waves, each wave owns a 32x32 tile; block covers 32 rows x 128 cols.
// Optionally also writes a transposed bf16 copy (CoutT[n*ldT + m]).
// ---------------------------------------------------------------------------
template <bool TANH, bool OUT_BF16, bool WRITE_T>
__global__ __launch_bounds__(128) void k_gemm_wmma(
    const bf16_t* __restrict__ A, const bf16_t* __restrict__ W,
    const float* __restrict__ bias, void* __restrict__ Cout,
    bf16_t* __restrict__ CoutT, int ldT, int M, int N, int K) {
  int lane = threadIdx.x & 31;
  int w = threadIdx.x >> 5;
  int n0 = blockIdx.x * 128 + w * 32;
  int m0 = blockIdx.y * 32;
  (void)M;
  v8f acc[2][2];
#pragma unroll
  for (int i = 0; i < 2; ++i)
#pragma unroll
    for (int j = 0; j < 2; ++j) acc[i][j] = zero8();

  for (int k0 = 0; k0 < K; k0 += 32) {
    v16bf a0 = load_a_frag(A + (size_t)m0 * K + k0, K, lane);
    v16bf a1 = load_a_frag(A + (size_t)(m0 + 16) * K + k0, K, lane);
    v16bf b0 = load_b_frag_rows(W + (size_t)n0 * K + k0, K, lane);
    v16bf b1 = load_b_frag_rows(W + (size_t)(n0 + 16) * K + k0, K, lane);
    acc[0][0] = wmma_bf16(a0, b0, acc[0][0]);
    acc[0][1] = wmma_bf16(a0, b1, acc[0][1]);
    acc[1][0] = wmma_bf16(a1, b0, acc[1][0]);
    acc[1][1] = wmma_bf16(a1, b1, acc[1][1]);
  }

  int hi = lane >> 4, nl = lane & 15;
#pragma unroll
  for (int mi = 0; mi < 2; ++mi)
#pragma unroll
    for (int ni = 0; ni < 2; ++ni) {
      int n = n0 + ni * 16 + nl;
      float bv = bias ? bias[n] : 0.f;
#pragma unroll
      for (int v = 0; v < 8; ++v) {
        int m = m0 + mi * 16 + hi * 8 + v;
        float val = acc[mi][ni][v] + bv;
        if (TANH) val = tanhf(val);
        if (OUT_BF16)
          ((bf16_t*)Cout)[(size_t)m * N + n] = f2bf(val);
        else
          ((float*)Cout)[(size_t)m * N + n] = val;
        if (WRITE_T) CoutT[(size_t)n * ldT + m] = f2bf(val);
      }
    }
}

// ---------------------------------------------------------------------------
// GRU scan: one persistent workgroup, 1024 threads = 32 waves for L2 latency
// hiding. Thread (u, khalf) accumulates half the K range of the three gate
// dot products for hidden unit u from the bf16-packed weights (one b64 load
// per k). Partials combined through LDS each step; h lives in LDS.
// Writes q (last TGT states) as bf16 directly into outb[:, 0:HID].
// ---------------------------------------------------------------------------
__global__ __launch_bounds__(1024, 1) void k_gru_scan(
    const float* __restrict__ gi, const bf16_t* __restrict__ wpk,
    const float* __restrict__ b_hh, bf16_t* __restrict__ outb) {
  __shared__ float h[HID];
  __shared__ float part[1024 * 3];
  int tid = threadIdx.x;
  int u = tid & (HID - 1);
  int kh = tid >> 9;  // 0 or 1: K half
  if (tid < HID) h[tid] = 0.f;
  float br = 0.f, bz = 0.f, bn = 0.f;
  if (tid < HID) {
    br = b_hh[u];
    bz = b_hh[HID + u];
    bn = b_hh[2 * HID + u];
  }
  __syncthreads();

  const v4bf* wbase = (const v4bf*)wpk + (size_t)(kh * 256) * HID + u;
  const float* hb = h + kh * 256;

  for (int t = 0; t < SEQ; ++t) {
    float gr = 0.f, gz = 0.f, gn = 0.f;
    for (int k4 = 0; k4 < 64; ++k4) {
      v4f hk = *(const v4f*)(hb + k4 * 4);
#pragma unroll
      for (int i = 0; i < 4; ++i) {
        v4bf wv = wbase[(size_t)(k4 * 4 + i) * HID];
        gr += (float)wv[0] * hk[i];
        gz += (float)wv[1] * hk[i];
        gn += (float)wv[2] * hk[i];
      }
    }
    part[tid * 3 + 0] = gr;
    part[tid * 3 + 1] = gz;
    part[tid * 3 + 2] = gn;
    __syncthreads();  // all h reads + partials done
    if (tid < HID) {
      const float* git = gi + (size_t)t * (3 * HID);
      float Gr = part[u * 3 + 0] + part[(u + HID) * 3 + 0] + br;
      float Gz = part[u * 3 + 1] + part[(u + HID) * 3 + 1] + bz;
      float Gn = part[u * 3 + 2] + part[(u + HID) * 3 + 2] + bn;
      float r = 1.f / (1.f + expf(-(git[u] + Gr)));
      float z = 1.f / (1.f + expf(-(git[HID + u] + Gz)));
      float n = tanhf(git[2 * HID + u] + r * Gn);
      float hnew = (1.f - z) * n + z * h[u];
      h[u] = hnew;
      if (t >= SEQ - TGT) outb[(size_t)(t - (SEQ - TGT)) * OUTD + u] = f2bf(hnew);
    }
    __syncthreads();  // h updated before next step reads it
  }
}

// ---------------------------------------------------------------------------
// Flash attention: context = softmax(q @ H^T) @ H, H = 32768 x 512 bf16.
// WG = 8 waves (256 threads), 32 q-rows per WG, 32-row H chunks in LDS,
// double-buffered via the Tensor Data Mover (wave 0 issues tensor_load_to_lds
// for chunk c+1 while all waves compute chunk c; s_wait_tensorcnt + barrier).
//   QK: split-K — wave (tw,kh) computes a half-K partial of one 16x16 S tile.
//   PV: each wave owns 64 context columns; B-frags from the f-major HsT slice.
// ---------------------------------------------------------------------------
static constexpr int LSTR = 520;  // bf16 row stride for Qs/Hs in LDS
static constexpr int TSTR = 40;   // bf16 row stride for HsT (f-major) in LDS
static constexpr int PSTR = 32;

__global__ __launch_bounds__(256, 1) void k_flash_attn(
    const bf16_t* __restrict__ Hb, const bf16_t* __restrict__ HbT,
    bf16_t* __restrict__ outb) {
  extern __shared__ char smem[];
  bf16_t* Qs   = (bf16_t*)smem;                  // 32 x LSTR
  bf16_t* Hs0  = Qs + 32 * LSTR;                 // 32 x LSTR (row-major chunk)
  bf16_t* Hs1  = Hs0 + 32 * LSTR;
  bf16_t* HsT0 = Hs1 + 32 * LSTR;                // 512 x TSTR (f-major chunk)
  bf16_t* HsT1 = HsT0 + 512 * TSTR;
  float*  Sbuf = (float*)(HsT1 + 512 * TSTR);    // 2 x 32 x 32 split-K partials
  bf16_t* Pb   = (bf16_t*)(Sbuf + 2 * 32 * 32);  // 32 x 32
  float*  m_s  = (float*)(Pb + 32 * 32);
  float*  l_s  = m_s + 32;
  float*  fac  = l_s + 32;

  int tid = threadIdx.x, lane = tid & 31, w = tid >> 5;
  int q0 = blockIdx.x * 32;

  // load the 32x512 Q block (bf16 q lives in outb[:,0:512])
  for (int idx = tid; idx < 32 * 64; idx += 256) {
    int r = idx >> 6, seg = idx & 63;
    *(v8bf*)(Qs + r * LSTR + seg * 8) =
        *(const v8bf*)(outb + (size_t)(q0 + r) * OUTD + seg * 8);
  }
  if (tid < 32) { m_s[tid] = -3.0e38f; l_s[tid] = 0.f; }

  v8f acc[2][4];
#pragma unroll
  for (int rt = 0; rt < 2; ++rt)
#pragma unroll
    for (int ct = 0; ct < 4; ++ct) acc[rt][ct] = zero8();

#if FLASH_TDM
  if (w == 0) {  // DMA chunk 0 into buffer 0, wait for it
    tdm_load_2d((unsigned)(unsigned long long)Hs0, Hb, 512, 32, 512, 7u, 4u);
    tdm_load_2d((unsigned)(unsigned long long)HsT0, HbT, 32, 512, HIS, 3u, 4u);
    __builtin_amdgcn_s_wait_tensorcnt(0);
  }
#endif
  __syncthreads();

  int buf = 0;
  for (int c = 0; c < HIS; c += 32) {
    bf16_t* Hcur = buf ? Hs1 : Hs0;
    bf16_t* Tcur = buf ? HsT1 : HsT0;
#if FLASH_TDM
    if (w == 0 && c + 32 < HIS) {  // async DMA of next chunk into other buffer
      bf16_t* Hnxt = buf ? Hs0 : Hs1;
      bf16_t* Tnxt = buf ? HsT0 : HsT1;
      tdm_load_2d((unsigned)(unsigned long long)Hnxt,
                  Hb + (size_t)(c + 32) * HID, 512, 32, 512, 7u, 4u);
      tdm_load_2d((unsigned)(unsigned long long)Tnxt,
                  HbT + (c + 32), 32, 512, HIS, 3u, 4u);
    }
#else
    {  // batched synchronous staging (8 loads in flight before the stores)
      v8bf t0[8], t1[8];
#pragma unroll
      for (int i = 0; i < 8; ++i) {
        int idx = tid + i * 256, r = idx >> 6, seg = idx & 63;
        t0[i] = *(const v8bf*)(Hb + (size_t)(c + r) * HID + seg * 8);
      }
#pragma unroll
      for (int i = 0; i < 8; ++i) {
        int idx = tid + i * 256, f = idx >> 2, sg = idx & 3;
        t1[i] = *(const v8bf*)(HbT + (size_t)f * HIS + c + sg * 8);
      }
#pragma unroll
      for (int i = 0; i < 8; ++i) {
        int idx = tid + i * 256, r = idx >> 6, seg = idx & 63;
        *(v8bf*)(Hcur + r * LSTR + seg * 8) = t0[i];
      }
#pragma unroll
      for (int i = 0; i < 8; ++i) {
        int idx = tid + i * 256, f = idx >> 2, sg = idx & 3;
        *(v8bf*)(Tcur + f * TSTR + sg * 8) = t1[i];
      }
      if (c + 32 < HIS) __builtin_prefetch(Hb + (size_t)(c + 32) * HID + tid * 128, 0, 0);
    }
    __syncthreads();
#endif

    // ---- QK: S(32x32) = Q(32x512) @ Hchunk^T, split-K over wave pairs ----
    {
      int tw = w & 3, kh = w >> 2;
      int wr = tw >> 1, wc = tw & 1;
      v8f s = zero8();
      for (int kk = kh * 256; kk < kh * 256 + 256; kk += 32) {
        v16bf a = load_a_frag(Qs + wr * 16 * LSTR + kk, LSTR, lane);
        v16bf b = load_b_frag_rows(Hcur + wc * 16 * LSTR + kk, LSTR, lane);
        s = wmma_bf16(a, b, s);
      }
      int hi = lane >> 4, nl = lane & 15;
#pragma unroll
      for (int v = 0; v < 8; ++v)
        Sbuf[kh * 1024 + (wr * 16 + hi * 8 + v) * 32 + wc * 16 + nl] = s[v];
    }
    __syncthreads();

    // ---- online softmax (one thread per q row, sums the two K partials) ----
    if (tid < 32) {
      int r = tid;
      float mo = m_s[r], mc = mo;
      float sv[32];
#pragma unroll
      for (int jc = 0; jc < 32; ++jc) {
        sv[jc] = Sbuf[r * 32 + jc] + Sbuf[1024 + r * 32 + jc];
        mc = fmaxf(mc, sv[jc]);
      }
      float scale = expf(mo - mc);
      float ls = 0.f;
#pragma unroll
      for (int jc = 0; jc < 32; ++jc) {
        float p = expf(sv[jc] - mc);
        ls += p;
        Pb[r * PSTR + jc] = f2bf(p);
      }
      l_s[r] = l_s[r] * scale + ls;
      m_s[r] = mc;
      fac[r] = scale;
    }
    __syncthreads();

    // ---- rescale accumulators, context += P(32x32) @ Hchunk(32x512) ----
    {
      int hi = lane >> 4;
#pragma unroll
      for (int rt = 0; rt < 2; ++rt) {
        float f[8];
#pragma unroll
        for (int v = 0; v < 8; ++v) f[v] = fac[rt * 16 + hi * 8 + v];
#pragma unroll
        for (int ct = 0; ct < 4; ++ct)
#pragma unroll
          for (int v = 0; v < 8; ++v) acc[rt][ct][v] *= f[v];
        v16bf a = load_a_frag(Pb + rt * 16 * PSTR, PSTR, lane);
#pragma unroll
        for (int ct = 0; ct < 4; ++ct) {
          v16bf b = load_b_frag_rows(Tcur + (w * 64 + ct * 16) * TSTR, TSTR, lane);
          acc[rt][ct] = wmma_bf16(a, b, acc[rt][ct]);
        }
      }
    }
    __syncthreads();
#if FLASH_TDM
    if (w == 0) __builtin_amdgcn_s_wait_tensorcnt(0);  // next buffer landed
    __syncthreads();
    buf ^= 1;
#endif
  }

  // epilogue: divide by l, write bf16 context columns
  int hi = lane >> 4, nl = lane & 15;
#pragma unroll
  for (int rt = 0; rt < 2; ++rt)
#pragma unroll
    for (int ct = 0; ct < 4; ++ct)
#pragma unroll
      for (int v = 0; v < 8; ++v) {
        int m = rt * 16 + hi * 8 + v;
        int col = HID + w * 64 + ct * 16 + nl;
        float val = acc[rt][ct][v] / l_s[m];
        outb[(size_t)(q0 + m) * OUTD + col] = f2bf(val);
      }
}

// ---------------------------------------------------------------------------
// Row-wise log_softmax, in place (2048 rows x 512)
// ---------------------------------------------------------------------------
__global__ void k_log_softmax(float* __restrict__ y, int N) {
  __shared__ float red[256];
  int row = blockIdx.x, tid = threadIdx.x;
  float* yr = y + (size_t)row * N;
  float mx = -3.0e38f;
  for (int j = tid; j < N; j += 256) mx = fmaxf(mx, yr[j]);
  red[tid] = mx;
  __syncthreads();
  for (int s = 128; s > 0; s >>= 1) {
    if (tid < s) red[tid] = fmaxf(red[tid], red[tid + s]);
    __syncthreads();
  }
  mx = red[0];
  __syncthreads();
  float sum = 0.f;
  for (int j = tid; j < N; j += 256) sum += expf(yr[j] - mx);
  red[tid] = sum;
  __syncthreads();
  for (int s = 128; s > 0; s >>= 1) {
    if (tid < s) red[tid] += red[tid + s];
    __syncthreads();
  }
  float lse = mx + logf(red[0]);
  __syncthreads();
  for (int j = tid; j < N; j += 256) yr[j] = yr[j] - lse;
}

// ---------------------------------------------------------------------------
// Launch
// ---------------------------------------------------------------------------
extern "C" void kernel_launch(void* const* d_in, const int* in_sizes, int n_in,
                              void* d_out, int out_size, void* d_ws, size_t ws_size,
                              hipStream_t stream) {
  (void)in_sizes; (void)n_in; (void)out_size; (void)ws_size;
  const float* loc   = (const float*)d_in[0];
  const float* tim   = (const float*)d_in[1];
  const float* hloc  = (const float*)d_in[2];
  const float* htim  = (const float*)d_in[3];
  const int*   uid   = (const int*)d_in[5];
  const float* fa_w  = (const float*)d_in[8];
  const float* fa_b  = (const float*)d_in[9];
  const float* w_ih  = (const float*)d_in[10];
  const float* w_hh  = (const float*)d_in[11];
  const float* b_ih  = (const float*)d_in[12];
  const float* b_hh  = (const float*)d_in[13];
  const float* e_uid = (const float*)d_in[14];
  const float* ff_w  = (const float*)d_in[15];
  const float* ff_b  = (const float*)d_in[16];
  float* out = (float*)d_out;

  size_t off = 0;
  auto carve = [&](size_t bytes) {
    size_t p = off;
    off = (off + bytes + 255) & ~(size_t)255;
    return p;
  };
  char* ws = (char*)d_ws;
  bf16_t* xb    = (bf16_t*)(ws + carve((size_t)SEQ * IN * 2));
  bf16_t* histb = (bf16_t*)(ws + carve((size_t)HIS * IN * 2));
  bf16_t* wab   = (bf16_t*)(ws + carve((size_t)HID * IN * 2));
  bf16_t* wihb  = (bf16_t*)(ws + carve((size_t)3 * HID * IN * 2));
  bf16_t* wfb   = (bf16_t*)(ws + carve((size_t)LOC_E * OUTD * 2));
  bf16_t* Hb    = (bf16_t*)(ws + carve((size_t)HIS * HID * 2));
  bf16_t* HbT   = (bf16_t*)(ws + carve((size_t)HID * HIS * 2));
  float*  gi    = (float*)(ws + carve((size_t)SEQ * 3 * HID * 4));
  bf16_t* wpk   = (bf16_t*)(ws + carve((size_t)HID * HID * 4 * 2));
  bf16_t* outb  = (bf16_t*)(ws + carve((size_t)TGT * OUTD * 2));

  auto blocks = [](long n) { return (int)((n + 255) / 256); };

  // 1) concat + cast inputs to bf16, pack W_hh
  k_concat_cast<<<blocks((long)SEQ * IN), 256, 0, stream>>>(loc, tim, xb, SEQ, LOC_E, TIM_E);
  k_concat_cast<<<blocks((long)HIS * IN), 256, 0, stream>>>(hloc, htim, histb, HIS, LOC_E, TIM_E);
  k_cast<<<blocks((long)HID * IN), 256, 0, stream>>>(fa_w, wab, HID * IN);
  k_cast<<<blocks((long)3 * HID * IN), 256, 0, stream>>>(w_ih, wihb, 3 * HID * IN);
  k_cast<<<blocks((long)LOC_E * OUTD), 256, 0, stream>>>(ff_w, wfb, LOC_E * OUTD);
  k_pack_whh<<<blocks((long)HID * HID), 256, 0, stream>>>(w_hh, wpk);

  // 2) history = tanh(hist @ fc_attn_w^T + b) -> Hb (bf16) and HbT (f-major)
  k_gemm_wmma<true, true, true><<<dim3(HID / 128, HIS / 32), 128, 0, stream>>>(
      histb, wab, fa_b, Hb, HbT, HIS, HIS, HID, IN);

  // 3) gi = x @ w_ih^T + b_ih   (fp32 out)
  k_gemm_wmma<false, false, false><<<dim3(3 * HID / 128, SEQ / 32), 128, 0, stream>>>(
      xb, wihb, b_ih, gi, nullptr, 0, SEQ, 3 * HID, IN);

  // 4) serial GRU scan -> q (bf16) into outb[:, 0:512]
  k_gru_scan<<<1, 1024, 0, stream>>>(gi, wpk, b_hh, outb);

  // 5) uid embedding gather -> outb[:, 1024:1088]
  k_uid_embed<<<blocks((long)TGT * UID_E), 256, 0, stream>>>(uid, e_uid, outb);

  // 6) flash attention -> context into outb[:, 512:1024]
  size_t shmem = (size_t)3 * 32 * LSTR * 2 + (size_t)2 * 512 * TSTR * 2 +
                 2 * 32 * 32 * 4 + 32 * 32 * 2 + 3 * 32 * 4;
  k_flash_attn<<<TGT / 32, 256, shmem, stream>>>(Hb, HbT, outb);

  // 7) y = out @ fc_final_w^T + b  (fp32 straight into d_out)
  k_gemm_wmma<false, false, false><<<dim3(LOC_E / 128, TGT / 32), 128, 0, stream>>>(
      outb, wfb, ff_b, out, nullptr, 0, TGT, LOC_E, OUTD);

  // 8) log_softmax rows in place
  k_log_softmax<<<TGT, 256, 0, stream>>>(out, LOC_E);
}
